// EVEMixtralSparseBlock_4982162063460
// MI455X (gfx1250) — matrix operand order repaired and checked by reading, not compiled, not run on hardware
//
#include <hip/hip_runtime.h>
#include <hip/hip_bf16.h>

// ---------------------------------------------------------------------------
// Mixtral MoE block for gfx1250 (MI455X): bf16 WMMA, gathered top-2 routing,
// async global->LDS tile copies (ASYNCcnt) with double buffering.
// ---------------------------------------------------------------------------

typedef __attribute__((ext_vector_type(16))) __bf16 bf16x16;
typedef __attribute__((ext_vector_type(4)))  __bf16 bf16x4;
typedef __attribute__((ext_vector_type(2)))  __bf16 bf16x2;
typedef __attribute__((ext_vector_type(8)))  float  floatx8;
typedef __attribute__((ext_vector_type(4)))  float  floatx4;
typedef __attribute__((ext_vector_type(2)))  float  floatx2;
typedef int int4v __attribute__((vector_size(4 * sizeof(int))));

union FragAB { bf16x16 v; uint4 q[2]; };

#if defined(__gfx1250__) && \
    __has_builtin(__builtin_amdgcn_global_load_async_to_lds_b128) && \
    __has_builtin(__builtin_amdgcn_s_wait_asynccnt)
#define HAS_ASYNC 1
#else
#define HAS_ASYNC 0
#endif

// 16-byte global -> LDS copy. Async path: no VGPR staging, ASYNCcnt-tracked.
__device__ __forceinline__ void cp16(void* l, const void* g) {
#if HAS_ASYNC
  __builtin_amdgcn_global_load_async_to_lds_b128((int4v*)g, (int4v*)l, 0, 0);
#else
  *(uint4*)l = *(const uint4*)g;
#endif
}

__device__ __forceinline__ void wait_async_tiles(int pending) {
#if HAS_ASYNC
  if (pending) __builtin_amdgcn_s_wait_asynccnt(6);
  else         __builtin_amdgcn_s_wait_asynccnt(0);
#endif
}

// fp32 -> bf16 conversions expressed as type conversions so the backend can
// use native cvt instructions where available.
__device__ __forceinline__ unsigned short f2bf1(float f) {
  return __builtin_bit_cast(unsigned short, (__bf16)f);
}
__device__ __forceinline__ unsigned int packbf(float a, float b) {
  floatx2 fv = {a, b};
  bf16x2 bv = __builtin_convertvector(fv, bf16x2);
  return __builtin_bit_cast(unsigned int, bv);
}
__device__ __forceinline__ uint2 pack4(float4 f) {
  floatx4 fv = {f.x, f.y, f.z, f.w};
  bf16x4 bv = __builtin_convertvector(fv, bf16x4);
  return __builtin_bit_cast(uint2, bv);
}

// B fragment (16x16, K=h*16..h*16+15) from 16 consecutive fp32 in LDS.
__device__ __forceinline__ bf16x16 bfrag_from_f32(const float* p) {
  union { bf16x16 v; uint2 d[4]; } r;
  r.d[0] = pack4(*(const float4*)(p + 0));
  r.d[1] = pack4(*(const float4*)(p + 4));
  r.d[2] = pack4(*(const float4*)(p + 8));
  r.d[3] = pack4(*(const float4*)(p + 12));
  return r.v;
}

static constexpr int BK    = 32;   // K per WMMA step
static constexpr int LDSW  = 40;   // bf16 row stride (80B, 16B-aligned, conflict-free)
static constexpr int LDSWF = 36;   // fp32 row stride (144B, 16B-aligned, conflict-free)
static constexpr int BM13  = 128, BN13 = 64;
static constexpr int BM2   = 128, BN2  = 128;

// ---------------------------------------------------------------------------
__global__ void moe_init(float* __restrict__ out, int* __restrict__ idx,
                         float* __restrict__ wlist, int* __restrict__ counts,
                         int T, int E, int TH) {
  int g = blockIdx.x * blockDim.x + threadIdx.x;
  if (g < TH) out[g] = 0.0f;
  if (g < T) {
    idx[(size_t)E * T + g]   = g;
    wlist[(size_t)E * T + g] = 1.0f;
  }
  if (g < E) counts[g] = 0;
  if (g == E) counts[E] = T;
}

__global__ void moe_cvt(const float* __restrict__ x,
                        unsigned short* __restrict__ xb, int n4) {
  int g = blockIdx.x * blockDim.x + threadIdx.x;
  if (g < n4) {
    float4 f = ((const float4*)x)[g];
    ((uint2*)xb)[g] = pack4(f);
  }
}

// ---------------------------------------------------------------------------
// Router: one wave per token -> logits, softmax, top-2, gather lists.
// ---------------------------------------------------------------------------
__global__ __launch_bounds__(32) void moe_router(
    const float* __restrict__ x, const float* __restrict__ rw,
    const float* __restrict__ rb, float* __restrict__ logits,
    int* __restrict__ counts, int* __restrict__ idx,
    float* __restrict__ wlist, int T, int H) {
  const int t = blockIdx.x;
  const int lane = threadIdx.x;
  float acc[8] = {0.f,0.f,0.f,0.f,0.f,0.f,0.f,0.f};
  const float* xr = x + (size_t)t * H;
  for (int h = lane; h < H; h += 32) {
    float xv = xr[h];
#pragma unroll
    for (int e = 0; e < 8; ++e) acc[e] += xv * rw[e * H + h];
  }
#pragma unroll
  for (int off = 16; off > 0; off >>= 1) {
#pragma unroll
    for (int e = 0; e < 8; ++e) acc[e] += __shfl_xor(acc[e], off, 32);
  }
  if (lane == 0) {
    float mx = -1e30f;
#pragma unroll
    for (int e = 0; e < 8; ++e) {
      acc[e] += rb[e];
      logits[(size_t)t * 8 + e] = acc[e];
      mx = fmaxf(mx, acc[e]);
    }
    float p[8];
#pragma unroll
    for (int e = 0; e < 8; ++e) p[e] = __expf(acc[e] - mx);
    int i0 = 0;
#pragma unroll
    for (int e = 1; e < 8; ++e) if (p[e] > p[i0]) i0 = e;
    int i1 = (i0 == 0) ? 1 : 0;
#pragma unroll
    for (int e = 0; e < 8; ++e) if (e != i0 && p[e] > p[i1]) i1 = e;
    float s  = p[i0] + p[i1];
    float g0 = p[i0] / s, g1 = p[i1] / s;
    int s0 = atomicAdd(&counts[i0], 1);
    idx[(size_t)i0 * T + s0]   = t;
    wlist[(size_t)i0 * T + s0] = g0;
    int s1 = atomicAdd(&counts[i1], 1);
    idx[(size_t)i1 * T + s1]   = t;
    wlist[(size_t)i1 * T + s1] = g1;
  }
}

// ---------------------------------------------------------------------------
// GEMM13: g[i,n] = silu(x[tok_i].w1[n]) * (x[tok_i].w3[n]) -> bf16 gbuf
// 128x64 block, 8 waves (4 over M, 2 over N), each wave 32x32 for BOTH mats.
// ---------------------------------------------------------------------------
__global__ __launch_bounds__(256) void moe_gemm13(
    const unsigned short* __restrict__ xb,   // [T,H] bf16
    const float* __restrict__ w1,            // [F,H] fp32
    const float* __restrict__ w3,            // [F,H] fp32
    const int* __restrict__ idx, const int* __restrict__ cnt,
    unsigned short* __restrict__ gbuf,       // [T,F] bf16, list-ordered rows
    int H, int F) {
  const int count = *cnt;
  const int m0 = blockIdx.x * BM13;
  if (m0 >= count) return;
  const int n0 = blockIdx.y * BN13;

  __shared__ __align__(16) unsigned short sA[2][BM13 * LDSW];
  __shared__ __align__(16) float sB1[2][BN13 * LDSWF];
  __shared__ __align__(16) float sB3[2][BN13 * LDSWF];
  __shared__ int sIdx[BM13];

  const int tid = threadIdx.x;
  if (tid < BM13) {
    int i = m0 + tid;
    sIdx[tid] = idx[(i < count) ? i : (count - 1)];
  }
  __syncthreads();

  // ---- per-thread tile-copy addresses (loop-invariant, registers) ----
  const int arow0 = tid >> 2, aseg = tid & 3, arow1 = arow0 + 64;
  const unsigned short* aG0 = xb + (size_t)sIdx[arow0] * H + aseg * 8;
  const unsigned short* aG1 = xb + (size_t)sIdx[arow1] * H + aseg * 8;
  const int aO0 = arow0 * LDSW + aseg * 8;
  const int aO1 = arow1 * LDSW + aseg * 8;
  const int brow0 = tid >> 3, bseg = tid & 7, brow1 = brow0 + 32;
  const float* b1G0 = w1 + (size_t)(n0 + brow0) * H + bseg * 4;
  const float* b1G1 = w1 + (size_t)(n0 + brow1) * H + bseg * 4;
  const float* b3G0 = w3 + (size_t)(n0 + brow0) * H + bseg * 4;
  const float* b3G1 = w3 + (size_t)(n0 + brow1) * H + bseg * 4;
  const int bO0 = brow0 * LDSWF + bseg * 4;
  const int bO1 = brow1 * LDSWF + bseg * 4;

  auto load_tiles = [&](int buf, int k0) {
    cp16(&sA[buf][aO0],  aG0 + k0);
    cp16(&sA[buf][aO1],  aG1 + k0);
    cp16(&sB1[buf][bO0], b1G0 + k0);
    cp16(&sB1[buf][bO1], b1G1 + k0);
    cp16(&sB3[buf][bO0], b3G0 + k0);
    cp16(&sB3[buf][bO1], b3G1 + k0);
  };

  floatx8 acc1[2][2] = {};
  floatx8 acc3[2][2] = {};
  const int lane = tid & 31, wid = tid >> 5;
  const int wm = wid & 3, wn = wid >> 2;
  const int lhalf = lane >> 4, lmod = lane & 15;

  const int nk = H / BK;
  load_tiles(0, 0);                                   // prime buffer 0
  for (int it = 0; it < nk; ++it) {
    const int cur = it & 1;
    __syncthreads();                                  // buf[cur^1] reusable
    const int more = (it + 1 < nk);
    if (more) load_tiles(cur ^ 1, (it + 1) * BK);     // prefetch next tile
    wait_async_tiles(more);                           // my buf[cur] copies done
    __syncthreads();                                  // everyone's done

    FragAB a[2];
    bf16x16 b1[2], b3[2];
#pragma unroll
    for (int mt = 0; mt < 2; ++mt) {
      const unsigned short* p = &sA[cur][(wm * 32 + mt * 16 + lmod) * LDSW + lhalf * 16];
      a[mt].q[0] = *(const uint4*)(p);
      a[mt].q[1] = *(const uint4*)(p + 8);
    }
#pragma unroll
    for (int nt = 0; nt < 2; ++nt) {
      const int c = wn * 32 + nt * 16 + lmod;
      b1[nt] = bfrag_from_f32(&sB1[cur][c * LDSWF + lhalf * 16]);
      b3[nt] = bfrag_from_f32(&sB3[cur][c * LDSWF + lhalf * 16]);
    }
#pragma unroll
    for (int mt = 0; mt < 2; ++mt) {
#pragma unroll
      for (int nt = 0; nt < 2; ++nt) {
        acc1[mt][nt] = __builtin_amdgcn_wmma_f32_16x16x32_bf16(
            false, a[mt].v, false, b1[nt], (short)0, acc1[mt][nt], false, false);
        acc3[mt][nt] = __builtin_amdgcn_wmma_f32_16x16x32_bf16(
            false, a[mt].v, false, b3[nt], (short)0, acc3[mt][nt], false, false);
      }
    }
  }

  // Epilogue: g = silu(h1) * h3 -> bf16
#pragma unroll
  for (int mt = 0; mt < 2; ++mt) {
#pragma unroll
    for (int nt = 0; nt < 2; ++nt) {
#pragma unroll
      for (int r = 0; r < 8; ++r) {
        int m = wm * 32 + mt * 16 + lhalf * 8 + r;
        int i = m0 + m;
        if (i < count) {
          int n = n0 + wn * 32 + nt * 16 + lmod;
          float h1 = acc1[mt][nt][r];
          float h3 = acc3[mt][nt][r];
          float g  = (h1 / (1.0f + __expf(-h1))) * h3;
          gbuf[(size_t)i * F + n] = f2bf1(g);
        }
      }
    }
  }
}

// ---------------------------------------------------------------------------
// GEMM2: y = g . w2^T ; out[tok_i,n] += gate_i * y[i,n]
// 128x128 block, 8 waves (2 over M, 4 over N), each wave 64x32.
// ---------------------------------------------------------------------------
__global__ __launch_bounds__(256) void moe_gemm2(
    const unsigned short* __restrict__ gbuf,  // [T,F] bf16 list-ordered
    const float* __restrict__ w2,             // [H,F] fp32
    const int* __restrict__ idx, const float* __restrict__ wlist,
    const int* __restrict__ cnt, float* __restrict__ out, int H, int F) {
  const int count = *cnt;
  const int m0 = blockIdx.x * BM2;
  if (m0 >= count) return;
  const int n0 = blockIdx.y * BN2;

  __shared__ __align__(16) unsigned short sA[2][BM2 * LDSW];
  __shared__ __align__(16) float sB[2][BN2 * LDSWF];
  __shared__ int   sTok[BM2];
  __shared__ float sW[BM2];

  const int tid = threadIdx.x;
  if (tid < BM2) {
    int i  = m0 + tid;
    int ic = (i < count) ? i : (count - 1);
    sTok[tid] = idx[ic];
    sW[tid]   = wlist[ic];
  }

  // ---- per-thread copy addresses ----
  const int arow0 = tid >> 2, aseg = tid & 3, arow1 = arow0 + 64;
  int ga0 = m0 + arow0; if (ga0 >= count) ga0 = count - 1;
  int ga1 = m0 + arow1; if (ga1 >= count) ga1 = count - 1;
  const unsigned short* aG0 = gbuf + (size_t)ga0 * F + aseg * 8;
  const unsigned short* aG1 = gbuf + (size_t)ga1 * F + aseg * 8;
  const int aO0 = arow0 * LDSW + aseg * 8;
  const int aO1 = arow1 * LDSW + aseg * 8;

  const int brow = tid >> 3, bseg = tid & 7;   // B: 1024 segs -> 4 per thread
  const float* bG[4];
  int bO[4];
#pragma unroll
  for (int j = 0; j < 4; ++j) {
    int row = brow + j * 32;
    bG[j] = w2 + (size_t)(n0 + row) * F + bseg * 4;
    bO[j] = row * LDSWF + bseg * 4;
  }

  auto load_tiles = [&](int buf, int k0) {
    cp16(&sA[buf][aO0], aG0 + k0);
    cp16(&sA[buf][aO1], aG1 + k0);
#pragma unroll
    for (int j = 0; j < 4; ++j) cp16(&sB[buf][bO[j]], bG[j] + k0);
  };

  floatx8 acc[4][2] = {};
  const int lane = tid & 31, wid = tid >> 5;
  const int wm = wid & 1, wn = wid >> 1;
  const int lhalf = lane >> 4, lmod = lane & 15;

  const int nk = F / BK;
  load_tiles(0, 0);
  for (int it = 0; it < nk; ++it) {
    const int cur = it & 1;
    __syncthreads();
    const int more = (it + 1 < nk);
    if (more) load_tiles(cur ^ 1, (it + 1) * BK);
    wait_async_tiles(more);
    __syncthreads();

    FragAB a[4];
    bf16x16 b[2];
#pragma unroll
    for (int mt = 0; mt < 4; ++mt) {
      const unsigned short* p = &sA[cur][(wm * 64 + mt * 16 + lmod) * LDSW + lhalf * 16];
      a[mt].q[0] = *(const uint4*)(p);
      a[mt].q[1] = *(const uint4*)(p + 8);
    }
#pragma unroll
    for (int nt = 0; nt < 2; ++nt) {
      const int c = wn * 32 + nt * 16 + lmod;
      b[nt] = bfrag_from_f32(&sB[cur][c * LDSWF + lhalf * 16]);
    }
#pragma unroll
    for (int mt = 0; mt < 4; ++mt) {
#pragma unroll
      for (int nt = 0; nt < 2; ++nt) {
        acc[mt][nt] = __builtin_amdgcn_wmma_f32_16x16x32_bf16(
            false, a[mt].v, false, b[nt], (short)0, acc[mt][nt], false, false);
      }
    }
  }

  // Scatter-accumulate (one writer per (token,n) per expert launch)
#pragma unroll
  for (int mt = 0; mt < 4; ++mt) {
#pragma unroll
    for (int nt = 0; nt < 2; ++nt) {
#pragma unroll
      for (int r = 0; r < 8; ++r) {
        int m = wm * 64 + mt * 16 + lhalf * 8 + r;
        int i = m0 + m;
        if (i < count) {
          int tok = sTok[m];
          float g = sW[m];
          int n = n0 + wn * 32 + nt * 16 + lmod;
          out[(size_t)tok * H + n] += g * acc[mt][nt][r];
        }
      }
    }
  }
}

// ---------------------------------------------------------------------------
// Host launcher
// ---------------------------------------------------------------------------
extern "C" void kernel_launch(void* const* d_in, const int* in_sizes, int n_in,
                              void* d_out, int out_size, void* d_ws, size_t ws_size,
                              hipStream_t stream) {
  const float* x   = (const float*)d_in[0];
  const float* rw  = (const float*)d_in[1];
  const float* rb  = (const float*)d_in[2];
  const float* w1  = (const float*)d_in[3];
  const float* w2  = (const float*)d_in[4];
  const float* w3  = (const float*)d_in[5];
  const float* sw1 = (const float*)d_in[6];
  const float* sw2 = (const float*)d_in[7];
  const float* sw3 = (const float*)d_in[8];

  const int E = in_sizes[2];                 // 8
  const int H = in_sizes[1] / E;             // 2048
  const int T = in_sizes[0] / H;             // 8192
  const int F = in_sizes[3] / (E * H);       // 4096

  float* out    = (float*)d_out;
  float* logits = out + (size_t)T * H;

  // Workspace: xb | gbuf | idx | wlist | counts  (~101 MB)
  char* ws = (char*)d_ws;
  unsigned short* xb   = (unsigned short*)ws;  ws += (size_t)T * H * 2;
  unsigned short* gbuf = (unsigned short*)ws;  ws += (size_t)T * F * 2;
  int*   idx   = (int*)ws;                     ws += (size_t)(E + 1) * T * 4;
  float* wlist = (float*)ws;                   ws += (size_t)(E + 1) * T * 4;
  int*   counts = (int*)ws;

  const int TH = T * H;

  moe_init<<<(TH + 255) / 256, 256, 0, stream>>>(out, idx, wlist, counts, T, E, TH);
  moe_cvt<<<((TH / 4) + 255) / 256, 256, 0, stream>>>(x, xb, TH / 4);
  moe_router<<<T, 32, 0, stream>>>(x, rw, rb, logits, counts, idx, wlist, T, H);

  for (int e = 0; e <= E; ++e) {
    const float* pw1 = (e < E) ? (w1 + (size_t)e * F * H) : sw1;
    const float* pw3 = (e < E) ? (w3 + (size_t)e * F * H) : sw3;
    const float* pw2 = (e < E) ? (w2 + (size_t)e * H * F) : sw2;
    dim3 g13(T / BM13, F / BN13);
    moe_gemm13<<<g13, 256, 0, stream>>>(xb, pw1, pw3, idx + (size_t)e * T,
                                        counts + e, gbuf, H, F);
    dim3 g2(T / BM2, H / BN2);
    moe_gemm2<<<g2, 256, 0, stream>>>(gbuf, pw2, idx + (size_t)e * T,
                                      wlist + (size_t)e * T, counts + e, out, H, F);
  }
}
